// AttentionPETL_57054345560268
// MI455X (gfx1250) — matrix operand order repaired and checked by reading, not compiled, not run on hardware
//
#include <hip/hip_runtime.h>
#include <hip/hip_bf16.h>

// ---------------------------------------------------------------------------
// MHA forward for B=8, N=1024, C=768, H=12, HD=64 on gfx1250 (MI455X).
// All GEMMs use v_wmma_f32_16x16x32_bf16 (bf16 inputs, fp32 accumulate).
// Attention is flash-style (no [B,H,N,N] materialization).
// V is stored transposed [B,H,HD,N] so the P*V B-fragments load with the
// reduction dim contiguous (coalescable to global_load_b128, like K).
// float->bf16 uses the native hardware convert (RNE), not bit emulation.
// ---------------------------------------------------------------------------

typedef __bf16 bf16_t;
typedef __attribute__((ext_vector_type(16))) __bf16        v16bf;
typedef __attribute__((ext_vector_type(4)))  __bf16        v4bf;
typedef __attribute__((ext_vector_type(8)))  float         v8f;
typedef __attribute__((ext_vector_type(8)))  unsigned int  v8u;

#define B_   8
#define N_   1024
#define C_   768
#define H_   12
#define HD_  64
#define TOK_ (B_ * N_)   // 8192

union FragU { v16bf bf; v8u u; };

// float -> bf16 via native hardware conversion (round-to-nearest-even)
__device__ __forceinline__ bf16_t f2b(float f) {
    return (bf16_t)f;
}

__device__ __forceinline__ v8f wmma_bf16(v16bf a, v16bf b, v8f c) {
    // (neg_a, A, neg_b, B, c_mod, C, reuse_a, reuse_b)
    return __builtin_amdgcn_wmma_f32_16x16x32_bf16(false, a, false, b,
                                                   (short)0, c, false, false);
}

// A-fragment: 16x32 bf16 tile, row-major source, leading dim ld (elements).
// ISA layout: lane m=lane&15, h=lane>>4; VGPR j<4 -> K=2j+8h, j>=4 -> 16+2(j-4)+8h.
// Pairs contiguous in K -> 32-bit loads (compiler coalesces to b128 clauses).
__device__ __forceinline__ v16bf load_frag_a(const bf16_t* __restrict__ src,
                                             int ld, int lane) {
    const int m = lane & 15, h = lane >> 4;
    const unsigned* p = reinterpret_cast<const unsigned*>(src);
    FragU f;
#pragma unroll
    for (int j = 0; j < 8; ++j) {
        int kk = (j < 4) ? (2 * j + 8 * h) : (16 + 2 * (j - 4) + 8 * h);
        f.u[j] = p[(m * ld + kk) >> 1];
    }
    return f.bf;
}

// B-fragment (32x16), element B(k,n) = w[n*ld + k] (reduction dim contiguous).
// Layout: lane n=lane&15, h=lane>>4; VGPR j<4 -> k = 2j + 8h, j>=4 -> 16 + 2(j-4) + 8h.
__device__ __forceinline__ v16bf load_frag_b_kmajor(const bf16_t* __restrict__ w,
                                                    int ld, int lane) {
    const int n = lane & 15, h = lane >> 4;
    const unsigned* p = reinterpret_cast<const unsigned*>(w);
    FragU f;
#pragma unroll
    for (int j = 0; j < 8; ++j) {
        int k = (j < 4) ? (2 * j + 8 * h) : (16 + 2 * (j - 4) + 8 * h);
        f.u[j] = p[(n * ld + k) >> 1];
    }
    return f.bf;
}

// ---------------------------------------------------------------------------
// Kernel 0: fp32 -> bf16 conversion (4 elements / thread, b64 packed store)
// ---------------------------------------------------------------------------
__global__ void cvt_f32_bf16(const float* __restrict__ s, bf16_t* __restrict__ d,
                             int n4) {
    int i = blockIdx.x * blockDim.x + threadIdx.x;
    if (i < n4) {
        float4 f = reinterpret_cast<const float4*>(s)[i];
        v4bf o = { f2b(f.x), f2b(f.y), f2b(f.z), f2b(f.w) };
        *reinterpret_cast<v4bf*>(d + 4 * (size_t)i) = o;
    }
}

// ---------------------------------------------------------------------------
// Kernel 1: QKV projection. qkv[t, dd] = sum_c x[t,c] * w_qkv[dd,c]
// 256 threads = 8 waves arranged 2(M) x 4(N); each wave owns a 32x64 tile
// (2x4 accumulators, 8 WMMA per K-step from 2 A + 4 B fragment loads).
// Scatter: Q,K -> [B][H][N][64]; V -> transposed [B][H][64][N].
// grid = (TOK/64, 2304/256)
// ---------------------------------------------------------------------------
__global__ void qkv_gemm(const bf16_t* __restrict__ xb,
                         const bf16_t* __restrict__ wb,
                         bf16_t* __restrict__ qb,
                         bf16_t* __restrict__ kb,
                         bf16_t* __restrict__ vb) {
    const int lane = threadIdx.x & 31;
    const int wave = threadIdx.x >> 5;
    const int wm = wave >> 2, wn = wave & 3;
    const int m0 = blockIdx.x * 64 + wm * 32;
    const int n0 = blockIdx.y * 256 + wn * 64;

    v8f acc[2][4] = {};
#pragma unroll 2
    for (int k = 0; k < C_; k += 32) {
        v16bf a0 = load_frag_a(xb + (size_t)m0 * C_ + k, C_, lane);
        v16bf a1 = load_frag_a(xb + (size_t)(m0 + 16) * C_ + k, C_, lane);
        v16bf b0 = load_frag_b_kmajor(wb + (size_t)(n0 +  0) * C_ + k, C_, lane);
        v16bf b1 = load_frag_b_kmajor(wb + (size_t)(n0 + 16) * C_ + k, C_, lane);
        v16bf b2 = load_frag_b_kmajor(wb + (size_t)(n0 + 32) * C_ + k, C_, lane);
        v16bf b3 = load_frag_b_kmajor(wb + (size_t)(n0 + 48) * C_ + k, C_, lane);
        acc[0][0] = wmma_bf16(a0, b0, acc[0][0]);
        acc[0][1] = wmma_bf16(a0, b1, acc[0][1]);
        acc[0][2] = wmma_bf16(a0, b2, acc[0][2]);
        acc[0][3] = wmma_bf16(a0, b3, acc[0][3]);
        acc[1][0] = wmma_bf16(a1, b0, acc[1][0]);
        acc[1][1] = wmma_bf16(a1, b1, acc[1][1]);
        acc[1][2] = wmma_bf16(a1, b2, acc[1][2]);
        acc[1][3] = wmma_bf16(a1, b3, acc[1][3]);
    }

    const int cn = lane & 15, h = lane >> 4;
#pragma unroll
    for (int mi = 0; mi < 2; ++mi)
#pragma unroll
        for (int ni = 0; ni < 4; ++ni)
#pragma unroll
            for (int r = 0; r < 8; ++r) {
                int t  = m0 + mi * 16 + r + 8 * h;
                int dd = n0 + ni * 16 + cn;
                int s  = dd / C_;            // 0=q, 1=k, 2=v
                int hh = (dd >> 6) % H_;
                int hd = dd & (HD_ - 1);
                int b  = t >> 10;
                int n  = t & (N_ - 1);
                bf16_t* dst;
                size_t idx;
                if (s == 2) {   // V transposed: [B][H][HD][N]
                    dst = vb;
                    idx = (((size_t)b * H_ + hh) * HD_ + hd) * N_ + n;
                } else {        // Q/K: [B][H][N][HD]
                    dst = (s == 0) ? qb : kb;
                    idx = (((size_t)b * H_ + hh) * N_ + n) * HD_ + hd;
                }
                dst[idx] = f2b(acc[mi][ni][r]);
            }
}

// ---------------------------------------------------------------------------
// Kernel 2: flash attention. One wave owns 16 query rows of one (b,h);
// streams keys/values 32 at a time. 8 waves/block -> 128 queries/block.
// grid = (B*H, N/128), block = 256.
// ---------------------------------------------------------------------------
__global__ void flash_attn(const bf16_t* __restrict__ q,
                           const bf16_t* __restrict__ k,
                           const bf16_t* __restrict__ vt,
                           bf16_t* __restrict__ o) {
    __shared__ bf16_t plds[8][16 * 32];   // wave-private P tile (row-major 16x32)

    const int lane = threadIdx.x & 31;
    const int wave = threadIdx.x >> 5;
    const int cn = lane & 15, h = lane >> 4;
    const int bh = blockIdx.x;                       // 0..95
    const int q0 = blockIdx.y * 128 + wave * 16;     // query row base

    const bf16_t* qh = q  + (size_t)bh * N_ * HD_;   // [N][64]
    const bf16_t* kh = k  + (size_t)bh * N_ * HD_;   // [N][64]
    const bf16_t* vh = vt + (size_t)bh * HD_ * N_;   // transposed [64][N]
    bf16_t* P = &plds[wave][0];

    // Q fragments for d=0..31 and d=32..63, kept resident.
    v16bf qa0 = load_frag_a(qh + (size_t)q0 * HD_ + 0, HD_, lane);
    v16bf qa1 = load_frag_a(qh + (size_t)q0 * HD_ + 32, HD_, lane);

    v8f acc[4] = {};                 // O tile: 16 x 64 (4 col-chunks of 16)
    float mrow[8], lrow[8];
#pragma unroll
    for (int r = 0; r < 8; ++r) { mrow[r] = -1e30f; lrow[r] = 0.f; }
    const float scale = 0.125f;      // HD^-0.5

    for (int j0 = 0; j0 < N_; j0 += 32) {
        // ---- S = Q * K^T for 32 keys (two 16x16 tiles) ----
        v8f s0 = {}, s1 = {};
        {
            v16bf b_lo = load_frag_b_kmajor(kh + (size_t)j0 * HD_ + 0, HD_, lane);
            v16bf b_hi = load_frag_b_kmajor(kh + (size_t)j0 * HD_ + 32, HD_, lane);
            s0 = wmma_bf16(qa0, b_lo, s0);
            s0 = wmma_bf16(qa1, b_hi, s0);
        }
        {
            v16bf b_lo = load_frag_b_kmajor(kh + (size_t)(j0 + 16) * HD_ + 0, HD_, lane);
            v16bf b_hi = load_frag_b_kmajor(kh + (size_t)(j0 + 16) * HD_ + 32, HD_, lane);
            s1 = wmma_bf16(qa0, b_lo, s1);
            s1 = wmma_bf16(qa1, b_hi, s1);
        }

        // ---- online softmax (fp32 stats; rows live in 16-lane groups) ----
        float corr[8];
#pragma unroll
        for (int r = 0; r < 8; ++r) {
            float a0 = s0[r] * scale;
            float a1 = s1[r] * scale;
            float rmax = fmaxf(a0, a1);
#pragma unroll
            for (int off = 1; off < 16; off <<= 1)
                rmax = fmaxf(rmax, __shfl_xor(rmax, off, 32));
            float mnew = fmaxf(mrow[r], rmax);
            float c = __expf(mrow[r] - mnew);
            mrow[r] = mnew;
            corr[r] = c;
            float p0 = __expf(a0 - mnew);
            float p1 = __expf(a1 - mnew);
            // write P (C-layout -> row-major LDS) for A-fragment reload
            P[(r + 8 * h) * 32 + cn]      = f2b(p0);
            P[(r + 8 * h) * 32 + cn + 16] = f2b(p1);
            float ls = p0 + p1;
#pragma unroll
            for (int off = 1; off < 16; off <<= 1)
                ls += __shfl_xor(ls, off, 32);
            lrow[r] = lrow[r] * c + ls;
        }
#pragma unroll
        for (int c4 = 0; c4 < 4; ++c4)
#pragma unroll
            for (int r = 0; r < 8; ++r)
                acc[c4][r] *= corr[r];

        // DS ops from this wave are in-order; make the P stores visible
        asm volatile("s_wait_dscnt 0" ::: "memory");

        // ---- O += P(16x32) * V(32x64); V^T gives k-contiguous B frags ----
        v16bf pa = load_frag_a(P, 32, lane);
#pragma unroll
        for (int c4 = 0; c4 < 4; ++c4) {
            v16bf vb = load_frag_b_kmajor(vh + (size_t)(c4 * 16) * N_ + j0,
                                          N_, lane);
            acc[c4] = wmma_bf16(pa, vb, acc[c4]);
        }
    }

    // ---- normalize + store to [B][N][C] bf16 ----
    const int b  = bh / H_;
    const int h0 = bh % H_;
#pragma unroll
    for (int c4 = 0; c4 < 4; ++c4)
#pragma unroll
        for (int r = 0; r < 8; ++r) {
            int row = q0 + r + 8 * h;
            int col = h0 * HD_ + c4 * 16 + cn;
            o[((size_t)b * N_ + row) * C_ + col] =
                f2b(acc[c4][r] / lrow[r]);
        }
}

// ---------------------------------------------------------------------------
// Kernel 3: output projection + bias.  out[t,d] = sum_c o[t,c]*w_proj[d,c]+b[d]
// Same 32x64-per-wave tiling as the QKV GEMM.
// grid = (TOK/64, 768/256), block = 256.
// ---------------------------------------------------------------------------
__global__ void proj_gemm(const bf16_t* __restrict__ ob,
                          const bf16_t* __restrict__ wb,
                          const float* __restrict__ bias,
                          float* __restrict__ out) {
    const int lane = threadIdx.x & 31;
    const int wave = threadIdx.x >> 5;
    const int wm = wave >> 2, wn = wave & 3;
    const int m0 = blockIdx.x * 64 + wm * 32;
    const int n0 = blockIdx.y * 256 + wn * 64;

    v8f acc[2][4] = {};
#pragma unroll 2
    for (int k = 0; k < C_; k += 32) {
        v16bf a0 = load_frag_a(ob + (size_t)m0 * C_ + k, C_, lane);
        v16bf a1 = load_frag_a(ob + (size_t)(m0 + 16) * C_ + k, C_, lane);
        v16bf b0 = load_frag_b_kmajor(wb + (size_t)(n0 +  0) * C_ + k, C_, lane);
        v16bf b1 = load_frag_b_kmajor(wb + (size_t)(n0 + 16) * C_ + k, C_, lane);
        v16bf b2 = load_frag_b_kmajor(wb + (size_t)(n0 + 32) * C_ + k, C_, lane);
        v16bf b3 = load_frag_b_kmajor(wb + (size_t)(n0 + 48) * C_ + k, C_, lane);
        acc[0][0] = wmma_bf16(a0, b0, acc[0][0]);
        acc[0][1] = wmma_bf16(a0, b1, acc[0][1]);
        acc[0][2] = wmma_bf16(a0, b2, acc[0][2]);
        acc[0][3] = wmma_bf16(a0, b3, acc[0][3]);
        acc[1][0] = wmma_bf16(a1, b0, acc[1][0]);
        acc[1][1] = wmma_bf16(a1, b1, acc[1][1]);
        acc[1][2] = wmma_bf16(a1, b2, acc[1][2]);
        acc[1][3] = wmma_bf16(a1, b3, acc[1][3]);
    }

    const int cn = lane & 15, h = lane >> 4;
#pragma unroll
    for (int mi = 0; mi < 2; ++mi)
#pragma unroll
        for (int ni = 0; ni < 4; ++ni)
#pragma unroll
            for (int r = 0; r < 8; ++r) {
                int t = m0 + mi * 16 + r + 8 * h;
                int d = n0 + ni * 16 + cn;
                out[(size_t)t * C_ + d] = acc[mi][ni][r] + bias[d];
            }
}

// ---------------------------------------------------------------------------
extern "C" void kernel_launch(void* const* d_in, const int* in_sizes, int n_in,
                              void* d_out, int out_size, void* d_ws, size_t ws_size,
                              hipStream_t stream) {
    (void)in_sizes; (void)n_in; (void)out_size; (void)ws_size;

    const float* x      = (const float*)d_in[0];   // [B,N,C]
    const float* w_qkv  = (const float*)d_in[1];   // [3C,C]
    const float* w_proj = (const float*)d_in[2];   // [C,C]
    const float* b_proj = (const float*)d_in[3];   // [C]
    float* out = (float*)d_out;                    // [B,N,C] fp32

    // workspace carve-up (bf16 buffers), 256B aligned
    char*  ws  = (char*)d_ws;
    size_t off = 0;
    auto carve = [&](size_t bytes) {
        void* p = ws + off;
        off += (bytes + 255) & ~size_t(255);
        return p;
    };
    bf16_t* xb     = (bf16_t*)carve((size_t)TOK_ * C_ * 2);
    bf16_t* wqkvb  = (bf16_t*)carve((size_t)3 * C_ * C_ * 2);
    bf16_t* wprojb = (bf16_t*)carve((size_t)C_ * C_ * 2);
    bf16_t* qb     = (bf16_t*)carve((size_t)B_ * H_ * N_ * HD_ * 2);
    bf16_t* kb     = (bf16_t*)carve((size_t)B_ * H_ * N_ * HD_ * 2);
    bf16_t* vtb    = (bf16_t*)carve((size_t)B_ * H_ * HD_ * N_ * 2);
    bf16_t* obuf   = (bf16_t*)carve((size_t)TOK_ * C_ * 2);

    // 0) fp32 -> bf16 conversions
    {
        int n4 = TOK_ * C_ / 4;
        cvt_f32_bf16<<<(n4 + 255) / 256, 256, 0, stream>>>(x, xb, n4);
        n4 = 3 * C_ * C_ / 4;
        cvt_f32_bf16<<<(n4 + 255) / 256, 256, 0, stream>>>(w_qkv, wqkvb, n4);
        n4 = C_ * C_ / 4;
        cvt_f32_bf16<<<(n4 + 255) / 256, 256, 0, stream>>>(w_proj, wprojb, n4);
    }

    // 1) QKV projection -> per-head Q/K (row-major) and V (transposed)
    {
        dim3 grid(TOK_ / 64, (3 * C_) / 256);
        qkv_gemm<<<grid, 256, 0, stream>>>(xb, wqkvb, qb, kb, vtb);
    }

    // 2) flash attention -> obuf [B,N,C] bf16
    {
        dim3 grid(B_ * H_, N_ / 128);
        flash_attn<<<grid, 256, 0, stream>>>(qb, kb, vtb, obuf);
    }

    // 3) output projection + bias -> d_out fp32
    {
        dim3 grid(TOK_ / 64, C_ / 256);
        proj_gemm<<<grid, 256, 0, stream>>>(obuf, wprojb, b_proj, out);
    }
}
